// FlexAttention_40699110097361
// MI455X (gfx1250) — matrix operand order repaired and checked by reading, not compiled, not run on hardware
//
#include <hip/hip_runtime.h>
#include <hip/hip_bf16.h>

typedef __attribute__((ext_vector_type(16))) _Float16 v16h;
typedef __attribute__((ext_vector_type(8)))  _Float16 v8h;
typedef __attribute__((ext_vector_type(4)))  _Float16 v4h;
typedef __attribute__((ext_vector_type(8)))  float    v8f;
typedef __attribute__((ext_vector_type(4)))  unsigned int u32x4;
typedef __attribute__((ext_vector_type(8)))  int      i32x8;
typedef __attribute__((ext_vector_type(4)))  int      i32x4;

#define H_   16
#define N_   2048
#define D_   1024
#define HD_  64
#define B_   2
#define SCALE_ 0.125f

// D = A(16x32 f16) * B(32x16 f16) + C(16x16 f32), wave32
static __device__ __forceinline__ v8f wmma16x16x32(v16h a, v16h b, v8f c) {
  return __builtin_amdgcn_wmma_f32_16x16x32_f16(
      /*neg_a=*/false, a, /*neg_b=*/false, b,
      /*c_mod=*/(short)0, c, /*reuse_a=*/false, /*reuse_b=*/false);
}

// Assemble an A-fragment (16x32, M=lane&15, K-base = (lane>>4)?16:0) from two
// contiguous 8-half chunks: VGPRs 0..3 hold K kb+0..7, VGPRs 4..7 hold K kb+8..15.
static __device__ __forceinline__ v16h make_a(v8h lo, v8h hi2) {
  v16h a;
#pragma unroll
  for (int i = 0; i < 8; ++i) { a[i] = lo[i]; a[8 + i] = hi2[i]; }
  return a;
}

// ----------------------------------------------------------------- TDM load
// Issue a Tensor Data Mover load of a 64(rows) x 64(k, contiguous) f16 tile
// into LDS at lds_byte_off. Descriptor per CDNA5 ISA:
//   group0: count=1 | lds_addr | global_addr[56:0] | type=2 ("image")
//   group1: data_size=1 (2B), tensor_dim0/1, tile_dim0/1, tensor_dim0_stride
// Tracked by TENSORcnt; caller waits with s_wait_tensorcnt.
static __device__ __forceinline__ void tdm_load_w64x64(unsigned lds_byte_off,
                                                       const _Float16* g) {
  const unsigned long long ga = (unsigned long long)(uintptr_t)g;
  u32x4 g0;
  g0[0] = 1u;                                    // count=1, user descriptor
  g0[1] = lds_byte_off;                          // LDS dest (bytes)
  g0[2] = (unsigned)(ga & 0xFFFFFFFFu);          // global_addr lo
  g0[3] = (unsigned)((ga >> 32) & 0x1FFFFFFu) | (2u << 30);  // addr hi | type=2
  i32x8 g1;
  g1[0] = (int)(1u << 16);                       // data_size=1 -> 2 bytes
  g1[1] = (int)((D_ & 0xFFFFu) << 16);           // tensor_dim0 lo16
  g1[2] = (int)(((D_ >> 16) & 0xFFFFu) | ((D_ & 0xFFFFu) << 16)); // d0 hi|d1 lo
  g1[3] = (int)(((D_ >> 16) & 0xFFFFu) | (64u << 16));            // d1 hi|tile0
  g1[4] = 64;                                    // tile_dim1=64, tile_dim2=0
  g1[5] = D_;                                    // tensor_dim0_stride lo32
  g1[6] = 0;                                     // stride0 hi | stride1 lo
  g1[7] = 0;
  const i32x4 gz = {};
#if defined(__clang_major__) && (__clang_major__ >= 23)
  const i32x8 gz8 = {};
  __builtin_amdgcn_tensor_load_to_lds(g0, g1, gz, gz, gz8, 0);
#else
  __builtin_amdgcn_tensor_load_to_lds(g0, g1, gz, gz, 0);
#endif
}

// ---------------------------------------------------------------- cast kernel
__global__ void cast_f32_to_f16(const float* __restrict__ in,
                                _Float16* __restrict__ out, int n4) {
  int i = blockIdx.x * blockDim.x + threadIdx.x;
  if (i < n4) {
    const float4 v = ((const float4*)in)[i];
    v4h h;
    h[0] = (_Float16)v.x; h[1] = (_Float16)v.y;
    h[2] = (_Float16)v.z; h[3] = (_Float16)v.w;
    ((v4h*)out)[i] = h;
  }
}

// ------------------------------------------ QKV projection + per-head layernorm
// grid = (B*N/64, H, 3), block = 128 (4 waves). All 4 waves share one head's 64
// weight columns: wave0 TDM-loads W tiles (64x64 f16, double buffered) to LDS,
// all waves run 16x64 WMMA tiles against it. LayerNorm over HD=64 (q,k only).
// q,k stored [B,H,N,64]; v stored transposed [B,H,64,N] for the P*V B-frags.
__global__ __launch_bounds__(128) void proj_qkv_ln(
    const _Float16* __restrict__ xq, const _Float16* __restrict__ xk,
    const _Float16* __restrict__ xv,
    const _Float16* __restrict__ wq, const _Float16* __restrict__ wk,
    const _Float16* __restrict__ wv,
    _Float16* __restrict__ qh, _Float16* __restrict__ kh,
    _Float16* __restrict__ vT) {
  __shared__ __align__(128) _Float16 wtile[2][64][64];
  const int lane = threadIdx.x & 31;
  const int wave = threadIdx.x >> 5;
  const int l15  = lane & 15;
  const int hi   = lane >> 4;
  const int rowbase = blockIdx.x * 64 + wave * 16;
  const int h = blockIdx.y;
  const int t = blockIdx.z;
  const _Float16* x = (t == 0) ? xq : (t == 1) ? xk : xv;
  const _Float16* w = (t == 0) ? wq : (t == 1) ? wk : wv;
  const int colbase = h * HD_;
  const _Float16* wsrc = w + (size_t)colbase * D_;  // 64 weight rows, k from 0

  const unsigned lds0 = (unsigned)(uintptr_t)&wtile[0][0][0];
  const unsigned lds1 = (unsigned)(uintptr_t)&wtile[1][0][0];

  v8f acc[4] = {};
  const _Float16* xrow = x + (size_t)(rowbase + l15) * D_;

  // Prologue: stage chunk 0.
  if (wave == 0) {
    tdm_load_w64x64(lds0, wsrc);
    __builtin_amdgcn_s_wait_tensorcnt(0);
  }
  __syncthreads();

  for (int c = 0; c < D_ / 64; ++c) {
    // Prefetch next K-chunk via TDM while computing this one.
    if (wave == 0 && (c + 1) < D_ / 64)
      tdm_load_w64x64((c & 1) ? lds0 : lds1, wsrc + (c + 1) * 64);
    const _Float16(*wt)[64] = wtile[c & 1];
#pragma unroll
    for (int kk2 = 0; kk2 < 64; kk2 += 32) {
      const _Float16* pa = xrow + c * 64 + kk2 + hi * 16;
      v16h a = make_a(*(const v8h*)pa, *(const v8h*)(pa + 8));
#pragma unroll
      for (int j = 0; j < 4; ++j) {
        // B[k][n] = w[colbase+n][k]: 16 contiguous halves along K, from LDS.
        v16h bf = *(const v16h*)&wt[j * 16 + l15][kk2 + hi * 16];
        acc[j] = wmma16x16x32(a, bf, acc[j]);
      }
    }
    if (wave == 0) __builtin_amdgcn_s_wait_tensorcnt(0);
    __syncthreads();
  }

  if (t < 2) {  // q, k: layernorm over the 64 head columns, store [B,H,N,64]
    _Float16* y = (t == 0) ? qh : kh;
#pragma unroll
    for (int r = 0; r < 8; ++r) {
      float s = 0.f, q2 = 0.f;
#pragma unroll
      for (int j = 0; j < 4; ++j) { float v = acc[j][r]; s += v; q2 += v * v; }
#pragma unroll
      for (int msk = 1; msk < 16; msk <<= 1) {
        s  += __shfl_xor(s, msk, 32);
        q2 += __shfl_xor(q2, msk, 32);
      }
      const float mu   = s * (1.f / 64.f);
      const float var  = q2 * (1.f / 64.f) - mu * mu;
      const float rinv = rsqrtf(var + 1e-5f);
      const int row = 8 * hi + r;
      const int gflat = rowbase + row;
      const int bb = gflat >> 11;            // / N_
      const int nn = gflat & (N_ - 1);
      const size_t base = (((size_t)(bb * H_ + h)) * N_ + nn) * HD_;
#pragma unroll
      for (int j = 0; j < 4; ++j)
        y[base + j * 16 + l15] = (_Float16)((acc[j][r] - mu) * rinv);
    }
  } else {  // v: store transposed [B,H,64,N]
#pragma unroll
    for (int r = 0; r < 8; ++r) {
      const int row = 8 * hi + r;
      const int gflat = rowbase + row;
      const int bb = gflat >> 11;
      const int nn = gflat & (N_ - 1);
#pragma unroll
      for (int j = 0; j < 4; ++j) {
        const size_t base =
            (((size_t)(bb * H_ + h)) * HD_ + j * 16 + l15) * (size_t)N_ + nn;
        vT[base] = (_Float16)acc[j][r];
      }
    }
  }
}

// --------------------------------------------------------- flash attention
// grid = (N/64, H, B), block = 128 (4 waves). Each wave owns 16 query rows,
// streams 32-key tiles with online softmax; P goes C-layout -> LDS -> A-frag.
__global__ __launch_bounds__(128) void flash_attn(
    const _Float16* __restrict__ qh, const _Float16* __restrict__ kh,
    const _Float16* __restrict__ vT, _Float16* __restrict__ o16) {
  __shared__ __align__(32) _Float16 pl[4][16][32];
  const int lane = threadIdx.x & 31;
  const int wave = threadIdx.x >> 5;
  const int l15  = lane & 15;
  const int hi   = lane >> 4;
  const int h = blockIdx.y, b = blockIdx.z;
  const int qbase = blockIdx.x * 64 + wave * 16;
  const size_t bh = (size_t)(b * H_ + h);
  const _Float16* Q = qh + bh * (size_t)N_ * HD_;
  const _Float16* K = kh + bh * (size_t)N_ * HD_;
  const _Float16* V = vT + bh * (size_t)HD_ * N_;

  // Q A-frags for this wave's 16 rows, K-dim split 0..31 / 32..63 (registers).
  v16h aq[2];
#pragma unroll
  for (int dh = 0; dh < 2; ++dh) {
    const _Float16* p = Q + (size_t)(qbase + l15) * HD_ + dh * 32 + hi * 16;
    aq[dh] = make_a(*(const v8h*)p, *(const v8h*)(p + 8));
  }

  v8f O[4] = {};
  float mrun[8], lrun[8];
#pragma unroll
  for (int r = 0; r < 8; ++r) { mrun[r] = -1e30f; lrun[r] = 0.f; }

  for (int kt = 0; kt < N_; kt += 32) {
    if (kt + 32 < N_) {
      __builtin_prefetch(K + (size_t)(kt + 32 + lane) * HD_, 0, 0);
      __builtin_prefetch(V + (size_t)(lane * 2) * N_ + kt + 32, 0, 0);
    }
    // S = Q * K^T * SCALE for two 16-key sub-tiles.
    v8f S[2] = {};
#pragma unroll
    for (int ks = 0; ks < 2; ++ks) {
      const int key = kt + ks * 16 + l15;
#pragma unroll
      for (int dh = 0; dh < 2; ++dh) {
        const _Float16* pb = K + (size_t)key * HD_ + dh * 32 + hi * 16;
        v16h bk = *(const v16h*)pb;
        S[ks] = wmma16x16x32(aq[dh], bk, S[ks]);
      }
    }
    // Online softmax: lanes 0-15 own rows 0-7, lanes 16-31 own rows 8-15.
#pragma unroll
    for (int r = 0; r < 8; ++r) {
      const float v0 = S[0][r] * SCALE_;
      const float v1 = S[1][r] * SCALE_;
      float mx = fmaxf(v0, v1);
#pragma unroll
      for (int msk = 1; msk < 16; msk <<= 1)
        mx = fmaxf(mx, __shfl_xor(mx, msk, 32));
      const float mnew = fmaxf(mrun[r], mx);
      const float p0 = __expf(v0 - mnew);
      const float p1 = __expf(v1 - mnew);
      float rs = p0 + p1;
#pragma unroll
      for (int msk = 1; msk < 16; msk <<= 1) rs += __shfl_xor(rs, msk, 32);
      const float resc = __expf(mrun[r] - mnew);
      mrun[r] = mnew;
      lrun[r] = lrun[r] * resc + rs;
#pragma unroll
      for (int j = 0; j < 4; ++j) O[j][r] *= resc;
      pl[wave][8 * hi + r][l15]      = (_Float16)p0;
      pl[wave][8 * hi + r][16 + l15] = (_Float16)p1;
    }
    // P (16x32) as A-fragment via LDS layout conversion (same-wave, DScnt-ordered).
    const _Float16* pp = &pl[wave][l15][hi * 16];
    v16h ap = make_a(*(const v8h*)pp, *(const v8h*)(pp + 8));
    // O += P * V  (V^T layout makes B-frags contiguous 32B loads).
#pragma unroll
    for (int j = 0; j < 4; ++j) {
      const _Float16* pv = V + (size_t)(j * 16 + l15) * N_ + kt + hi * 16;
      v16h bv = *(const v16h*)pv;
      O[j] = wmma16x16x32(ap, bv, O[j]);
    }
  }

  // Epilogue: normalize and store o in [B,N,D] f16 for the output projection.
#pragma unroll
  for (int r = 0; r < 8; ++r) {
    const float inv = 1.f / lrun[r];
    const int row = 8 * hi + r;
    const size_t base =
        ((size_t)(b * N_ + qbase + row)) * D_ + (size_t)h * HD_;
#pragma unroll
    for (int j = 0; j < 4; ++j)
      o16[base + j * 16 + l15] = (_Float16)(O[j][r] * inv);
  }
}

// ---------------------------------------------------------- output projection
// grid = (B*N/64, D/64), block = 128 (4 waves). Same TDM-staged weight tile
// pipeline as proj_qkv_ln. out = o @ wproj^T, f32 stores.
__global__ __launch_bounds__(128) void proj_out(
    const _Float16* __restrict__ o16, const _Float16* __restrict__ wp,
    float* __restrict__ out) {
  __shared__ __align__(128) _Float16 wtile[2][64][64];
  const int lane = threadIdx.x & 31;
  const int wave = threadIdx.x >> 5;
  const int l15  = lane & 15;
  const int hi   = lane >> 4;
  const int rowbase = blockIdx.x * 64 + wave * 16;
  const int colbase = blockIdx.y * 64;
  const _Float16* wsrc = wp + (size_t)colbase * D_;

  const unsigned lds0 = (unsigned)(uintptr_t)&wtile[0][0][0];
  const unsigned lds1 = (unsigned)(uintptr_t)&wtile[1][0][0];

  v8f acc[4] = {};
  const _Float16* xrow = o16 + (size_t)(rowbase + l15) * D_;

  if (wave == 0) {
    tdm_load_w64x64(lds0, wsrc);
    __builtin_amdgcn_s_wait_tensorcnt(0);
  }
  __syncthreads();

  for (int c = 0; c < D_ / 64; ++c) {
    if (wave == 0 && (c + 1) < D_ / 64)
      tdm_load_w64x64((c & 1) ? lds0 : lds1, wsrc + (c + 1) * 64);
    const _Float16(*wt)[64] = wtile[c & 1];
#pragma unroll
    for (int kk2 = 0; kk2 < 64; kk2 += 32) {
      const _Float16* pa = xrow + c * 64 + kk2 + hi * 16;
      v16h a = make_a(*(const v8h*)pa, *(const v8h*)(pa + 8));
#pragma unroll
      for (int j = 0; j < 4; ++j) {
        v16h bf = *(const v16h*)&wt[j * 16 + l15][kk2 + hi * 16];
        acc[j] = wmma16x16x32(a, bf, acc[j]);
      }
    }
    if (wave == 0) __builtin_amdgcn_s_wait_tensorcnt(0);
    __syncthreads();
  }

#pragma unroll
  for (int r = 0; r < 8; ++r) {
    const int row = 8 * hi + r;
    const size_t base = (size_t)(rowbase + row) * D_ + colbase;
#pragma unroll
    for (int j = 0; j < 4; ++j) out[base + j * 16 + l15] = acc[j][r];
  }
}

// ------------------------------------------------------------------- launcher
extern "C" void kernel_launch(void* const* d_in, const int* in_sizes, int n_in,
                              void* d_out, int out_size, void* d_ws,
                              size_t ws_size, hipStream_t stream) {
  (void)in_sizes; (void)n_in; (void)out_size; (void)ws_size;

  const float* xq = (const float*)d_in[0];
  const float* xk = (const float*)d_in[1];
  const float* xv = (const float*)d_in[2];
  const float* wq = (const float*)d_in[3];
  const float* wk = (const float*)d_in[4];
  const float* wv = (const float*)d_in[5];
  const float* wp = (const float*)d_in[6];
  float* out = (float*)d_out;

  // Workspace map (64 MB total; everything f16)
  char* ws = (char*)d_ws;
  _Float16* xq16 = (_Float16*)(ws + (size_t)0);
  _Float16* xk16 = (_Float16*)(ws + (size_t)(8u << 20));
  _Float16* xv16 = (_Float16*)(ws + (size_t)(16u << 20));
  _Float16* wq16 = (_Float16*)(ws + (size_t)(24u << 20));
  _Float16* wk16 = (_Float16*)(ws + (size_t)(26u << 20));
  _Float16* wv16 = (_Float16*)(ws + (size_t)(28u << 20));
  _Float16* wp16 = (_Float16*)(ws + (size_t)(30u << 20));
  _Float16* q16  = (_Float16*)(ws + (size_t)(32u << 20));  // [B,H,N,64]
  _Float16* k16  = (_Float16*)(ws + (size_t)(40u << 20));  // [B,H,N,64]
  _Float16* vT16 = (_Float16*)(ws + (size_t)(48u << 20));  // [B,H,64,N]
  _Float16* o16  = (_Float16*)(ws + (size_t)(56u << 20));  // [B,N,D]

  const int nx4 = (B_ * N_ * D_) / 4;   // 1,048,576
  const int nw4 = (D_ * D_) / 4;        // 262,144
  cast_f32_to_f16<<<nx4 / 256, 256, 0, stream>>>(xq, xq16, nx4);
  cast_f32_to_f16<<<nx4 / 256, 256, 0, stream>>>(xk, xk16, nx4);
  cast_f32_to_f16<<<nx4 / 256, 256, 0, stream>>>(xv, xv16, nx4);
  cast_f32_to_f16<<<nw4 / 256, 256, 0, stream>>>(wq, wq16, nw4);
  cast_f32_to_f16<<<nw4 / 256, 256, 0, stream>>>(wk, wk16, nw4);
  cast_f32_to_f16<<<nw4 / 256, 256, 0, stream>>>(wv, wv16, nw4);
  cast_f32_to_f16<<<nw4 / 256, 256, 0, stream>>>(wp, wp16, nw4);

  proj_qkv_ln<<<dim3((B_ * N_) / 64, H_, 3), 128, 0, stream>>>(
      xq16, xk16, xv16, wq16, wk16, wv16, q16, k16, vT16);

  flash_attn<<<dim3(N_ / 64, H_, B_), 128, 0, stream>>>(q16, k16, vT16, o16);

  proj_out<<<dim3((B_ * N_) / 64, D_ / 64), 128, 0, stream>>>(o16, wp16, out);
}
